// PhysBiformer_40853728920305
// MI455X (gfx1250) — matrix operand order repaired and checked by reading, not compiled, not run on hardware
//
#include <hip/hip_runtime.h>
#include <hip/hip_bf16.h>
#include <math.h>

typedef __bf16 bf16;
typedef __attribute__((ext_vector_type(16))) __bf16 v16bf;
typedef __attribute__((ext_vector_type(8)))  float  v8f;

#define L_TOK   10240
#define ROWS    81920           // T*b*L = 4*2*10240
#define BL_ROWS 20480           // b*L

__device__ __forceinline__ float bflo(unsigned int u) {
  union { unsigned int x; float f; } c; c.x = u << 16; return c.f;
}
__device__ __forceinline__ float bfhi(unsigned int u) {
  union { unsigned int x; float f; } c; c.x = u & 0xffff0000u; return c.f;
}

// ---------------- stem conv1: 3->16, k3, s4, p1 : out [2,16,40,16,16] ----------------
__global__ void k_stem1(const float* __restrict__ x, const float* __restrict__ w,
                        const float* __restrict__ b, float* __restrict__ out) {
  int i = blockIdx.x * blockDim.x + threadIdx.x;
  if (i >= 2*16*40*16*16) return;
  int ow = i & 15; int t = i >> 4;
  int oh = t & 15; t >>= 4;
  int od = t % 40; t /= 40;
  int oc = t & 15; int bb = t >> 4;
  float acc = b[oc];
  for (int ic = 0; ic < 3; ++ic)
    for (int kd = 0; kd < 3; ++kd) {
      int id = od*4 - 1 + kd; if (id < 0 || id >= 160) continue;
      for (int kh = 0; kh < 3; ++kh) {
        int ih = oh*4 - 1 + kh; if (ih < 0 || ih >= 64) continue;
        for (int kw = 0; kw < 3; ++kw) {
          int iw = ow*4 - 1 + kw; if (iw < 0 || iw >= 64) continue;
          acc += x[((bb*3+ic)*160 + id)*4096 + ih*64 + iw] *
                 w[(((oc*3+ic)*3+kd)*3+kh)*3+kw];
        }
      }
    }
  out[i] = acc;
}

// ---------------- stem conv2: 16->64, k3, s1, p1 : out [2,64,40,16,16] ----------------
__global__ void k_stem2(const float* __restrict__ xin, const float* __restrict__ w,
                        const float* __restrict__ b, float* __restrict__ out) {
  int i = blockIdx.x * blockDim.x + threadIdx.x;
  if (i >= 2*64*40*16*16) return;
  int ow = i & 15; int t = i >> 4;
  int oh = t & 15; t >>= 4;
  int od = t % 40; t /= 40;
  int oc = t & 63; int bb = t >> 6;
  float acc = b[oc];
  for (int ic = 0; ic < 16; ++ic)
    for (int kd = 0; kd < 3; ++kd) {
      int id = od - 1 + kd; if (id < 0 || id >= 40) continue;
      for (int kh = 0; kh < 3; ++kh) {
        int ih = oh - 1 + kh; if (ih < 0 || ih >= 16) continue;
        for (int kw = 0; kw < 3; ++kw) {
          int iw = ow - 1 + kw; if (iw < 0 || iw >= 16) continue;
          acc += xin[((bb*16+ic)*40 + id)*256 + ih*16 + iw] *
                 w[(((oc*16+ic)*3+kd)*3+kh)*3+kw];
        }
      }
    }
  out[i] = acc;
}

// ---------------- per-channel batch-stat BN for NCDHW (block per channel) ----------------
__global__ void k_bnstats_nc(const float* __restrict__ xin, int C, int SP,
                             float* __restrict__ mean, float* __restrict__ invstd) {
  int c = blockIdx.x;
  __shared__ float ssum[256], ssq[256];
  float s = 0.f, q = 0.f;
  for (int n = 0; n < 2; ++n) {
    const float* p = xin + (size_t)(n*C + c) * SP;
    for (int j = threadIdx.x; j < SP; j += 256) { float v = p[j]; s += v; q += v*v; }
  }
  ssum[threadIdx.x] = s; ssq[threadIdx.x] = q; __syncthreads();
  for (int st = 128; st > 0; st >>= 1) {
    if ((int)threadIdx.x < st) { ssum[threadIdx.x] += ssum[threadIdx.x+st];
                                 ssq[threadIdx.x]  += ssq[threadIdx.x+st]; }
    __syncthreads();
  }
  if (threadIdx.x == 0) {
    float cnt = 2.f * (float)SP;
    float m = ssum[0] / cnt;
    float v = ssq[0] / cnt - m*m;
    mean[c] = m; invstd[c] = rsqrtf(v + 1e-5f);
  }
}

__global__ void k_bn_relu(float* __restrict__ xio, int C, int SP, int total,
                          const float* __restrict__ mean, const float* __restrict__ invstd,
                          const float* __restrict__ g, const float* __restrict__ be) {
  int i = blockIdx.x * blockDim.x + threadIdx.x;
  if (i >= total) return;
  int c = (i / SP) % C;
  float v = (xio[i] - mean[c]) * invstd[c] * g[c] + be[c];
  xio[i] = v > 0.f ? v : 0.f;
}

// ---------------- tokenize + first LIF (constant input over T) ----------------
__global__ void k_tok_lif(const float* __restrict__ h2, float* __restrict__ xt) {
  int i = blockIdx.x * 256 + threadIdx.x;          // over b*L*C
  if (i >= BL_ROWS*64) return;
  int c = i & 63; int t2 = i >> 6;
  int l = t2 % L_TOK; int bb = t2 / L_TOK;
  float val = h2[(size_t)(bb*64 + c) * L_TOK + l];
  float v = 0.f;
  for (int t = 0; t < 4; ++t) {
    v = 0.5f * (v + val);
    float s = (v >= 1.f) ? 1.f : 0.f;
    v *= (1.f - s);
    xt[((size_t)(t*2 + bb) * L_TOK + l) * 64 + c] = s;
  }
}

// ---------------- block BN stats over [T,b,L] per channel (atomic partials) ----------------
__global__ void k_zero(float* __restrict__ p, int n) {
  int i = blockIdx.x * blockDim.x + threadIdx.x; if (i < n) p[i] = 0.f;
}

__global__ void k_bnstats_rows(const float* __restrict__ xt, float* __restrict__ acc) {
  __shared__ float ls[256], lq[256];
  int c = threadIdx.x & 63, g = threadIdx.x >> 6;
  float s = 0.f, q = 0.f;
  for (int r = blockIdx.x*4 + g; r < ROWS; r += gridDim.x*4) {
    float v = xt[(size_t)r*64 + c]; s += v; q += v*v;
  }
  ls[threadIdx.x] = s; lq[threadIdx.x] = q; __syncthreads();
  if (g == 0) {
    s = ls[c] + ls[64+c] + ls[128+c] + ls[192+c];
    q = lq[c] + lq[64+c] + lq[128+c] + lq[192+c];
    atomicAdd(&acc[c], s); atomicAdd(&acc[64+c], q);
  }
}

__global__ void k_bnfinal(const float* __restrict__ acc,
                          float* __restrict__ mean, float* __restrict__ invstd) {
  int c = threadIdx.x; if (c >= 64) return;
  float cnt = (float)ROWS;
  float m = acc[c] / cnt, v = acc[64+c] / cnt - m*m;
  mean[c] = m; invstd[c] = rsqrtf(v + 1e-5f);
}

// ---------------- BN + LIF over T -> bf16 spikes ----------------
__global__ void k_bn_lif(const float* __restrict__ xt, bf16* __restrict__ sbf,
                         const float* __restrict__ mean, const float* __restrict__ invstd,
                         const float* __restrict__ g, const float* __restrict__ be) {
  int i = blockIdx.x * blockDim.x + threadIdx.x;   // over b*L*C
  if (i >= BL_ROWS*64) return;
  int c = i & 63;
  float m = mean[c], is = invstd[c] * g[c], bt = be[c];
  float v = 0.f;
  for (int t = 0; t < 4; ++t) {
    size_t idx = (size_t)t * BL_ROWS * 64 + i;
    float xn = (xt[idx] - m) * is + bt;
    v = 0.5f * (v + xn);
    float s = (v >= 1.f) ? 1.f : 0.f;
    v *= (1.f - s);
    sbf[idx] = (bf16)s;
  }
}

// ---------------- pack f32 weight [K,N] into WMMA B-fragment order ----------------
// dst layout: [(kt*tiles_n + nt)*32 + lane][16 contiguous bf16]  (32B per lane)
__global__ void k_pack_b(const float* __restrict__ B, bf16* __restrict__ dst,
                         int K, int N) {
  int i = blockIdx.x * blockDim.x + threadIdx.x;
  if (i >= K * N) return;
  int e    = i & 15;
  int lane = (i >> 4) & 31;
  int tile = i >> 9;
  int tn   = N >> 4;
  int nt   = tile % tn;
  int kt   = tile / tn;
  int k    = kt*32 + ((lane >> 4) << 3) + (e & 7) + ((e >> 3) << 4);
  int nn   = nt*16 + (lane & 15);
  dst[i] = (bf16)B[(size_t)k * N + nn];
}

// ---------------- WMMA bf16 GEMM: C = A[M,K] @ B[K,N] + bias ----------------
// Wave computes 16x64 (1 M-tile x 4 N-tiles). A fragment is prefetched one
// k-step ahead; B fragments load as one clause at the top of each step.
// Live regs ~100 (32 acc + 16 A + 32 B) -> no spills.
__device__ __forceinline__ v16bf ldA(const bf16* p) {
  union { uint4 u[2]; v16bf v; } au;
  au.u[0] = *(const uint4*)(p);
  au.u[1] = *(const uint4*)(p + 16);
  return au.v;
}

template <int MODE>   // 0: bias->bf16   1: bias->GELU->bf16   2: out_f32 += scale*(acc+bias)
__global__ void __launch_bounds__(256)
k_wmma_gemm(const bf16* __restrict__ A, const bf16* __restrict__ Bp,
            const float* __restrict__ bias, void* __restrict__ Out,
            const float* __restrict__ scale, int M, int N, int K) {
  int lane = threadIdx.x & 31;
  int wave = threadIdx.x >> 5;
  int tiles_n = N >> 4;                       // multiple of 4
  int ngroups = tiles_n >> 2;
  int groups  = (M >> 4) * ngroups;
  int grp = blockIdx.x * 8 + wave;
  if (grp >= groups) return;                  // wave-uniform exit: EXEC all-ones
  int m0 = (grp / ngroups) << 4;
  int ng =  grp % ngroups;

  v8f acc0 = {0.f,0.f,0.f,0.f,0.f,0.f,0.f,0.f};
  v8f acc1 = acc0, acc2 = acc0, acc3 = acc0;

  const bf16* Arow = A + (size_t)(m0 + (lane & 15)) * K + ((lane >> 4) << 3);
  const v16bf* bp = (const v16bf*)Bp;
  size_t bstep = (size_t)tiles_n * 32;        // fragment index stride per k-step
  size_t bidx  = (size_t)(ng << 2) * 32 + lane;

  v16bf a = ldA(Arow);                        // prologue A fragment (k0 = 0)

  for (int k0 = 0; k0 < K; k0 += 32) {
    // current step's B fragments: 4 x 32B loads, single clause
    v16bf b0 = bp[bidx], b1 = bp[bidx+32], b2 = bp[bidx+64], b3 = bp[bidx+96];
    v16bf na;
    if (k0 + 32 < K) na = ldA(Arow + k0 + 32);   // prefetch next A (uniform branch)
    acc0 = __builtin_amdgcn_wmma_f32_16x16x32_bf16(false, a, false, b0, (short)0, acc0, false, false);
    acc1 = __builtin_amdgcn_wmma_f32_16x16x32_bf16(false, a, false, b1, (short)0, acc1, false, false);
    acc2 = __builtin_amdgcn_wmma_f32_16x16x32_bf16(false, a, false, b2, (short)0, acc2, false, false);
    acc3 = __builtin_amdgcn_wmma_f32_16x16x32_bf16(false, a, false, b3, (short)0, acc3, false, false);
    if (k0 + 32 < K) { a = na; bidx += bstep; }
  }

  int half = lane >> 4;
  int nn0  = (ng << 6) + (lane & 15);
  v8f accs[4] = {acc0, acc1, acc2, acc3};
#pragma unroll
  for (int t = 0; t < 4; ++t) {
    int nn = nn0 + t*16;
    float bv = bias[nn];
    if (MODE == 2) {
      float sc = scale[0];
      float* o = (float*)Out;
#pragma unroll
      for (int r = 0; r < 8; ++r) {
        size_t row = (size_t)(m0 + half*8 + r);
        o[row * N + nn] += sc * (accs[t][r] + bv);
      }
    } else {
      bf16* o = (bf16*)Out;
#pragma unroll
      for (int r = 0; r < 8; ++r) {
        size_t row = (size_t)(m0 + half*8 + r);
        float v = accs[t][r] + bv;
        if (MODE == 1) v = 0.5f * v * (1.f + erff(v * 0.70710678118654752f));
        o[row * N + nn] = (bf16)v;
      }
    }
  }
}

// ---------------- regional mean pooling of q,k per window ----------------
__global__ void k_regmean(const bf16* __restrict__ qkv,
                          float* __restrict__ qm, float* __restrict__ km) {
  int nw = blockIdx.x;                       // n*160 + w ; tokens nw*64 + j
  int c = threadIdx.x & 63, g = threadIdx.x >> 6;
  __shared__ float sq[256], sk[256];
  float aq = 0.f, ak = 0.f;
  const bf16* base = qkv + (size_t)nw * 64 * 192;
  for (int j = g; j < 64; j += 4) {
    const bf16* row = base + (size_t)j * 192;
    aq += (float)row[c];
    ak += (float)row[64 + c];
  }
  sq[threadIdx.x] = aq; sk[threadIdx.x] = ak; __syncthreads();
  if (g == 0) {
    aq = sq[c] + sq[64+c] + sq[128+c] + sq[192+c];
    ak = sk[c] + sk[64+c] + sk[128+c] + sk[192+c];
    qm[nw*64 + c] = aq * (1.f/64.f);
    km[nw*64 + c] = ak * (1.f/64.f);
  }
}

// ---------------- coarse routing: affinity + top-4 (stable ties like lax.top_k) ----------------
__global__ void k_route(const float* __restrict__ qm, const float* __restrict__ km,
                        int* __restrict__ idx) {
  int i = blockIdx.x * blockDim.x + threadIdx.x;   // over 8*160 query windows
  if (i >= 1280) return;
  int n = i / 160;
  const float* q = qm + (size_t)i * 64;
  float bv[4] = {-3.4e38f,-3.4e38f,-3.4e38f,-3.4e38f};
  int   bi[4] = {0,0,0,0};
  for (int j = 0; j < 160; ++j) {
    const float* k = km + (size_t)(n*160 + j) * 64;
    float a = 0.f;
#pragma unroll 16
    for (int c = 0; c < 64; ++c) a += q[c]*k[c];
    if (a > bv[3]) {
      int p = 3;
      while (p > 0 && a > bv[p-1]) { bv[p]=bv[p-1]; bi[p]=bi[p-1]; --p; }
      bv[p] = a; bi[p] = j;
    }
  }
  idx[i*4+0]=bi[0]; idx[i*4+1]=bi[1]; idx[i*4+2]=bi[2]; idx[i*4+3]=bi[3];
}

// ---------------- gathered attention: LDS-staged k/v + online softmax ----------------
__global__ void k_attn(const bf16* __restrict__ qkv, const int* __restrict__ idx,
                       bf16* __restrict__ obf) {
  __shared__ unsigned int kv[64][64];      // token j: [0..31]=k (32 uints), [32..63]=v
  int nw = blockIdx.x;
  int n = nw / 160;
  int head = threadIdx.x >> 6, row = threadIdx.x & 63;

  float q[16];
  {
    const unsigned int* qrow =
        (const unsigned int*)(qkv + ((size_t)(nw*64 + row)) * 192 + head*16);
#pragma unroll
    for (int c2 = 0; c2 < 8; ++c2) {
      unsigned int u = qrow[c2];
      q[2*c2]   = bflo(u);
      q[2*c2+1] = bfhi(u);
    }
  }

  float mx = -3.4e38f, l = 0.f, acc[16];
#pragma unroll
  for (int c = 0; c < 16; ++c) acc[c] = 0.f;

  int jld = threadIdx.x >> 2;              // token this thread helps load
  int pld = threadIdx.x & 3;               // 16-uint slice within token

  for (int rg = 0; rg < 4; ++rg) {
    int region = idx[nw*4 + rg];
    {
      const uint4* src = (const uint4*)(qkv +
          ((size_t)(n*L_TOK + region*64 + jld)) * 192 + 64);
      uint4* dstv = (uint4*)&kv[jld][pld*16];
#pragma unroll
      for (int qq = 0; qq < 4; ++qq) dstv[qq] = src[pld*4 + qq];
    }
    __syncthreads();

    for (int j = 0; j < 64; ++j) {
      const unsigned int* kk = &kv[j][head*8];
      float sc = 0.f;
#pragma unroll
      for (int c2 = 0; c2 < 8; ++c2) {
        unsigned int u = kk[c2];
        sc += q[2*c2]*bflo(u) + q[2*c2+1]*bfhi(u);
      }
      sc *= 0.25f;                         // hd^-0.5, hd=16
      float mnew = fmaxf(mx, sc);
      float corr = __expf(mx - mnew);
      float p    = __expf(sc - mnew);
      l = l * corr + p;
      const unsigned int* vv = &kv[j][32 + head*8];
#pragma unroll
      for (int c2 = 0; c2 < 8; ++c2) {
        unsigned int u = vv[c2];
        acc[2*c2]   = acc[2*c2]   * corr + p * bflo(u);
        acc[2*c2+1] = acc[2*c2+1] * corr + p * bfhi(u);
      }
      mx = mnew;
    }
    __syncthreads();
  }

  float inv = 1.f / l;
  bf16* orow = obf + ((size_t)(nw*64 + row)) * 64 + head*16;
#pragma unroll
  for (int c = 0; c < 16; ++c) orow[c] = (bf16)(acc[c] * inv);
}

// ---------------- final temporal + spatial mean : xm [2,40,64] ----------------
__global__ void k_final_mean(const float* __restrict__ xt, float* __restrict__ xm) {
  int bd = blockIdx.x;                     // b*40 + d
  int b = bd / 40, d = bd % 40;
  int c = threadIdx.x & 63, g = threadIdx.x >> 6;
  __shared__ float sm[256];
  float s = 0.f;
  for (int t = 0; t < 4; ++t) {
    const float* base = xt + ((size_t)(t*2 + b) * L_TOK + d*256) * 64;
    for (int sp = g; sp < 256; sp += 4) s += base[(size_t)sp*64 + c];
  }
  sm[threadIdx.x] = s; __syncthreads();
  if (g == 0) {
    s = sm[c] + sm[64+c] + sm[128+c] + sm[192+c];
    xm[bd*64 + c] = s * (1.f/1024.f);
  }
}

// ---------------- align_corners linear interp (40->160) + head ----------------
__global__ void k_head(const float* __restrict__ xm, const float* __restrict__ hw,
                       const float* __restrict__ hb, float* __restrict__ out) {
  int i = blockIdx.x * blockDim.x + threadIdx.x;   // 2*160
  if (i >= 320) return;
  int b = i / 160, f = i % 160;
  float pos = (float)f * (39.f / 159.f);
  int lo = (int)floorf(pos); if (lo > 39) lo = 39;
  int hi = lo + 1; if (hi > 39) hi = 39;
  float wf = pos - (float)lo;
  const float* a = xm + (size_t)(b*40 + lo) * 64;
  const float* c = xm + (size_t)(b*40 + hi) * 64;
  float r = 0.f;
  for (int ch = 0; ch < 64; ++ch) r += (a[ch]*(1.f-wf) + c[ch]*wf) * hw[ch];
  out[i] = r + hb[0];
}

// =====================================================================================
extern "C" void kernel_launch(void* const* d_in, const int* in_sizes, int n_in,
                              void* d_out, int out_size, void* d_ws, size_t ws_size,
                              hipStream_t stream) {
  (void)in_sizes; (void)n_in; (void)out_size; (void)ws_size;
  const float* x        = (const float*)d_in[0];
  const float* stem_w1  = (const float*)d_in[1];
  const float* stem_b1  = (const float*)d_in[2];
  const float* stem_g1  = (const float*)d_in[3];
  const float* stem_be1 = (const float*)d_in[4];
  const float* stem_w2  = (const float*)d_in[5];
  const float* stem_b2  = (const float*)d_in[6];
  const float* stem_g2  = (const float*)d_in[7];
  const float* stem_be2 = (const float*)d_in[8];
  const float* head_w   = (const float*)d_in[9];
  const float* head_b   = (const float*)d_in[10];

  // ---- workspace carve ----
  char* wp = (char*)d_ws;
  auto alloc = [&](size_t bytes) -> void* {
    void* p = (void*)wp; wp += (bytes + 255) & ~(size_t)255; return p;
  };
  float* h1     = (float*)alloc((size_t)327680 * 4);
  float* h2     = (float*)alloc((size_t)1310720 * 4);
  float* xt     = (float*)alloc((size_t)ROWS * 64 * 4);       // residual state [T,b,L,C]
  bf16*  sbf    = (bf16*) alloc((size_t)ROWS * 64 * 2);       // spikes
  bf16*  qkvbf  = (bf16*) alloc((size_t)ROWS * 192 * 2);
  bf16*  hbf    = (bf16*) alloc((size_t)ROWS * 256 * 2);      // FFN hidden
  bf16*  obf    = (bf16*) alloc((size_t)ROWS * 64 * 2);       // attention output
  float* qm     = (float*)alloc((size_t)1280 * 64 * 4);
  float* km     = (float*)alloc((size_t)1280 * 64 * 4);
  int*   idxbuf = (int*)  alloc((size_t)1280 * 4 * 4);
  float* xm     = (float*)alloc((size_t)2 * 40 * 64 * 4);
  float* meanC  = (float*)alloc(64 * 4);
  float* istdC  = (float*)alloc(64 * 4);
  float* accC   = (float*)alloc(128 * 4);
  bf16*  wqkv_p = (bf16*) alloc((size_t)64 * 192 * 2);        // packed B fragments
  bf16*  wo_p   = (bf16*) alloc((size_t)64 * 64 * 2);
  bf16*  w1_p   = (bf16*) alloc((size_t)64 * 256 * 2);
  bf16*  w2_p   = (bf16*) alloc((size_t)256 * 64 * 2);

  // ---- stem ----
  k_stem1<<<(327680 + 255) / 256, 256, 0, stream>>>(x, stem_w1, stem_b1, h1);
  k_bnstats_nc<<<16, 256, 0, stream>>>(h1, 16, 10240, meanC, istdC);
  k_bn_relu<<<(327680 + 255) / 256, 256, 0, stream>>>(h1, 16, 10240, 327680,
                                                      meanC, istdC, stem_g1, stem_be1);
  k_stem2<<<(1310720 + 255) / 256, 256, 0, stream>>>(h1, stem_w2, stem_b2, h2);
  k_bnstats_nc<<<64, 256, 0, stream>>>(h2, 64, 10240, meanC, istdC);
  k_bn_relu<<<(1310720 + 255) / 256, 256, 0, stream>>>(h2, 64, 10240, 1310720,
                                                       meanC, istdC, stem_g2, stem_be2);
  k_tok_lif<<<(BL_ROWS*64 + 255) / 256, 256, 0, stream>>>(h2, xt);

  // ---- transformer blocks ----
  for (int blk = 0; blk < 3; ++blk) {
    const int base = 11 + blk * 13;
    const float* g1    = (const float*)d_in[base + 0];
    const float* b1    = (const float*)d_in[base + 1];
    const float* g2    = (const float*)d_in[base + 2];
    const float* b2    = (const float*)d_in[base + 3];
    const float* wqkv  = (const float*)d_in[base + 4];
    const float* bqkv  = (const float*)d_in[base + 5];
    const float* wo    = (const float*)d_in[base + 6];
    const float* bo    = (const float*)d_in[base + 7];
    const float* wffn1 = (const float*)d_in[base + 8];
    const float* bffn1 = (const float*)d_in[base + 9];
    const float* wffn2 = (const float*)d_in[base + 10];
    const float* bffn2 = (const float*)d_in[base + 11];
    const float* scale = (const float*)d_in[base + 12];

    k_pack_b<<<(12288 + 255) / 256, 256, 0, stream>>>(wqkv,  wqkv_p, 64, 192);
    k_pack_b<<<(4096  + 255) / 256, 256, 0, stream>>>(wo,    wo_p,   64, 64);
    k_pack_b<<<(16384 + 255) / 256, 256, 0, stream>>>(wffn1, w1_p,   64, 256);
    k_pack_b<<<(16384 + 255) / 256, 256, 0, stream>>>(wffn2, w2_p,   256, 64);

    // --- attention half ---
    k_zero<<<1, 128, 0, stream>>>(accC, 128);
    k_bnstats_rows<<<320, 256, 0, stream>>>(xt, accC);
    k_bnfinal<<<1, 64, 0, stream>>>(accC, meanC, istdC);
    k_bn_lif<<<(BL_ROWS*64 + 255) / 256, 256, 0, stream>>>(xt, sbf, meanC, istdC, g1, b1);

    // qkv : [81920,64]@[64,192] ; groups = 5120 * 3
    k_wmma_gemm<0><<<1920, 256, 0, stream>>>(sbf, wqkv_p, bqkv, qkvbf, nullptr,
                                             ROWS, 192, 64);
    k_regmean<<<1280, 256, 0, stream>>>(qkvbf, qm, km);
    k_route<<<5, 256, 0, stream>>>(qm, km, idxbuf);
    k_attn<<<1280, 256, 0, stream>>>(qkvbf, idxbuf, obf);
    // wo + residual : xt += scale * (o @ wo + bo) ; groups = 5120
    k_wmma_gemm<2><<<640, 256, 0, stream>>>(obf, wo_p, bo, xt, scale, ROWS, 64, 64);

    // --- FFN half ---
    k_zero<<<1, 128, 0, stream>>>(accC, 128);
    k_bnstats_rows<<<320, 256, 0, stream>>>(xt, accC);
    k_bnfinal<<<1, 64, 0, stream>>>(accC, meanC, istdC);
    k_bn_lif<<<(BL_ROWS*64 + 255) / 256, 256, 0, stream>>>(xt, sbf, meanC, istdC, g2, b2);

    // ffn1 : groups = 5120 * 4 ; ffn2 : groups = 5120
    k_wmma_gemm<1><<<2560, 256, 0, stream>>>(sbf, w1_p, bffn1, hbf, nullptr,
                                             ROWS, 256, 64);
    k_wmma_gemm<2><<<640, 256, 0, stream>>>(hbf, w2_p, bffn2, xt, scale,
                                            ROWS, 64, 256);
  }

  // ---- pooling, interpolation, head ----
  k_final_mean<<<80, 256, 0, stream>>>(xt, xm);
  k_head<<<2, 160, 0, stream>>>(xm, head_w, head_b, (float*)d_out);
}
